// Net_57672820851006
// MI455X (gfx1250) — compile-verified, matrix operand order checked by reading, not run on hardware
//
#include <hip/hip_runtime.h>
#include <hip/hip_bf16.h>
#include <math.h>

typedef __attribute__((ext_vector_type(16))) _Float16 v16h;
typedef __attribute__((ext_vector_type(8)))  _Float16 h8;
typedef __attribute__((ext_vector_type(8)))  float    v8f;
typedef __attribute__((ext_vector_type(4)))  float    f4;

#define ACT_NONE 0
#define ACT_RELU 1
#define ACT_SIG  2
#define ACT_TANH 3

// ---------------------------------------------------------------- utilities
__global__ void cvt_f16_k(const float* __restrict__ in, _Float16* __restrict__ out, size_t n) {
  size_t i = (size_t)blockIdx.x * blockDim.x + threadIdx.x;
  if (i < n) out[i] = (_Float16)in[i];
}

__global__ void zero_f32_k(float* p, int n) {
  int i = blockIdx.x * blockDim.x + threadIdx.x;
  if (i < n) p[i] = 0.f;
}

__global__ void init_M_k(float* M, int B_) {
  int i = blockIdx.x * blockDim.x + threadIdx.x;
  if (i < B_ * 6) {
    int j = i % 6;
    M[i] = (j == 0 || j == 4) ? 1.f : 0.f;
  }
}

__device__ __forceinline__ v16h join8(h8 lo, h8 hi) {
  return __builtin_shufflevector(lo, hi, 0, 1, 2, 3, 4, 5, 6, 7,
                                          8, 9, 10, 11, 12, 13, 14, 15);
}

// ------------------------------------------- implicit-GEMM 3x3 SAME conv, WMMA
// in:  f16 NCHW activations     wgt: f32 OIHW (3x3)    out: f16 NCHW
// M = N*H*W pixels (tile 64), N = Cout (tile 64), K = Cin*9 (step 32)
// 8 waves: wave (wvM0, wvN) computes M-subtiles {wvM0, wvM0+2} x N-subtile wvN,
// sharing one B fragment across two WMMAs.
// LDS layouts are K-contiguous so每 fragment = 2 x ds_load_b128.
__global__ __launch_bounds__(256)
void conv3x3_wmma(const _Float16* __restrict__ in, const float* __restrict__ wgt,
                  const float* __restrict__ bias, _Float16* __restrict__ out,
                  int Nb, int Cin, int Cout, int H, int W, int relu)
{
  __shared__ __align__(16) _Float16 ldsA[64 * 32];    // [pixel][k]
  __shared__ __align__(16) _Float16 ldsBt[64 * 32];   // [cout][k]  (transposed)
  const int tid  = threadIdx.x;
  const int lane = tid & 31;
  const int wv   = tid >> 5;        // 0..7
  const int wvN  = wv & 3;          // N subtile
  const int wvM0 = wv >> 2;         // M subtiles wvM0 and wvM0+2
  const int HW   = H * W;
  const int Mtot = Nb * HW;
  const int pixBase = blockIdx.x * 64;
  const int chBase  = blockIdx.y * 64;
  const int Ktot = Cin * 9;

  const int m    = lane & 15;
  const int half = lane >> 4;
  const int rowA0 = (wvM0 * 16 + m) * 32;
  const int rowA1 = ((wvM0 + 2) * 16 + m) * 32;
  const int rowB  = (wvN * 16 + (lane & 15)) * 32;

  v8f acc0 = {0.f, 0.f, 0.f, 0.f, 0.f, 0.f, 0.f, 0.f};
  v8f acc1 = {0.f, 0.f, 0.f, 0.f, 0.f, 0.f, 0.f, 0.f};

  for (int kc = 0; kc < Ktot; kc += 32) {
    if (kc + 32 < Ktot)  // stream next weight chunk (global_prefetch_b8)
      __builtin_prefetch(&wgt[(size_t)chBase * Ktot + kc + 32], 0, 1);

    // A tile: 64 pixels x 32 k (implicit im2col gather)
    for (int e = tid; e < 64 * 32; e += 256) {
      int row = e >> 5, col = e & 31;
      int p = pixBase + row;
      int k = kc + col;
      _Float16 v = (_Float16)0.f;
      if (p < Mtot && k < Ktot) {
        int n = p / HW, rem = p % HW;
        int h = rem / W, w = rem % W;
        int cin = k / 9, r9 = k % 9;
        int kh = r9 / 3, kw = r9 % 3;
        int ih = h + kh - 1, iw = w + kw - 1;
        if (ih >= 0 && ih < H && iw >= 0 && iw < W)
          v = in[((size_t)(n * Cin + cin) * H + ih) * W + iw];
      }
      ldsA[row * 32 + col] = v;
    }
    // B tile (transposed): 64 cout x 32 k
    for (int e = tid; e < 64 * 32; e += 256) {
      int col = e >> 5, krow = e & 31;
      int k = kc + krow;
      int co = chBase + col;
      float v = 0.f;
      if (k < Ktot && co < Cout) {
        int cin = k / 9, r9 = k % 9;
        v = wgt[(size_t)(co * Cin + cin) * 9 + r9];
      }
      ldsBt[col * 32 + krow] = (_Float16)v;
    }
    __syncthreads();

    // fragments: two contiguous 8xf16 runs each -> ds_load_b128 pairs
    v16h af0 = join8(*(const h8*)&ldsA[rowA0 + half * 8],
                     *(const h8*)&ldsA[rowA0 + 16 + half * 8]);
    v16h af1 = join8(*(const h8*)&ldsA[rowA1 + half * 8],
                     *(const h8*)&ldsA[rowA1 + 16 + half * 8]);
    v16h bf  = join8(*(const h8*)&ldsBt[rowB + half * 8],
                     *(const h8*)&ldsBt[rowB + 16 + half * 8]);

    acc0 = __builtin_amdgcn_wmma_f32_16x16x32_f16(false, af0, false, bf,
                                                  (short)0, acc0, false, false);
    acc1 = __builtin_amdgcn_wmma_f32_16x16x32_f16(false, af1, false, bf,
                                                  (short)0, acc1, false, false);
    __syncthreads();
  }

  // store: C/D layout -> VGPR r holds M = r + 8*(lane>=16), N = lane&15
  int co = chBase + wvN * 16 + (lane & 15);
  if (co < Cout) {
    float bval = bias ? bias[co] : 0.f;
    #pragma unroll
    for (int r = 0; r < 8; ++r) {
      int p0 = pixBase + wvM0 * 16 + r + half * 8;
      if (p0 < Mtot) {
        int n = p0 / HW, rem = p0 % HW;
        float v = acc0[r] + bval;
        if (relu) v = fmaxf(v, 0.f);
        out[((size_t)(n * Cout + co)) * HW + rem] = (_Float16)v;
      }
      int p1 = pixBase + (wvM0 + 2) * 16 + r + half * 8;
      if (p1 < Mtot) {
        int n = p1 / HW, rem = p1 % HW;
        float v = acc1[r] + bval;
        if (relu) v = fmaxf(v, 0.f);
        out[((size_t)(n * Cout + co)) * HW + rem] = (_Float16)v;
      }
    }
  }
}

// ------------------------------------------------- small WMMA GEMM (M == 16)
// out(16 x N) = act( A(16 x K) @ W(N x K)^T + bias + addend )   all f32 I/O
// vectorized path: each lane's fragment = 4 x global_load_b128 + cvt
__global__ __launch_bounds__(32)
void gemm16_wmma(const float* __restrict__ A, const float* __restrict__ W,
                 const float* __restrict__ bias, const float* __restrict__ add,
                 float* __restrict__ out, int K, int Nc, int act)
{
  const int lane  = threadIdx.x & 31;
  const int nBase = blockIdx.x * 16;
  const int m     = lane & 15;
  const int half  = lane >> 4;
  const int n     = nBase + (lane & 15);
  const bool nval = (n < Nc);

  v8f acc = {0.f, 0.f, 0.f, 0.f, 0.f, 0.f, 0.f, 0.f};
  int kc = 0;
  for (; kc + 32 <= K; kc += 32) {
    if (kc + 32 < K) __builtin_prefetch(&A[(size_t)m * K + kc + 32], 0, 1);
    const float* ap = A + (size_t)m * K + kc + half * 8;
    f4 a0 = *(const f4*)ap;
    f4 a1 = *(const f4*)(ap + 4);
    f4 a2 = *(const f4*)(ap + 16);
    f4 a3 = *(const f4*)(ap + 20);
    f4 w0 = {0.f,0.f,0.f,0.f}, w1 = w0, w2 = w0, w3 = w0;
    if (nval) {
      const float* wp = W + (size_t)n * K + kc + half * 8;
      w0 = *(const f4*)wp;
      w1 = *(const f4*)(wp + 4);
      w2 = *(const f4*)(wp + 16);
      w3 = *(const f4*)(wp + 20);
    }
    v16h af, bf;
    #pragma unroll
    for (int i = 0; i < 4; ++i) {
      af[i]      = (_Float16)a0[i];  af[4 + i]  = (_Float16)a1[i];
      af[8 + i]  = (_Float16)a2[i];  af[12 + i] = (_Float16)a3[i];
      bf[i]      = (_Float16)w0[i];  bf[4 + i]  = (_Float16)w1[i];
      bf[8 + i]  = (_Float16)w2[i];  bf[12 + i] = (_Float16)w3[i];
    }
    acc = __builtin_amdgcn_wmma_f32_16x16x32_f16(false, af, false, bf,
                                                 (short)0, acc, false, false);
  }
  if (kc < K) {  // scalar tail (K % 32 != 0)
    v16h af, bf;
    #pragma unroll
    for (int j = 0; j < 8; ++j) {
      int k0 = kc + ((j >= 4) ? 16 : 0) + half * 8 + (j & 3) * 2;
      float a0 = (k0     < K) ? A[(size_t)m * K + k0]     : 0.f;
      float a1 = (k0 + 1 < K) ? A[(size_t)m * K + k0 + 1] : 0.f;
      float w0 = (nval && k0     < K) ? W[(size_t)n * K + k0]     : 0.f;
      float w1 = (nval && k0 + 1 < K) ? W[(size_t)n * K + k0 + 1] : 0.f;
      af[2 * j] = (_Float16)a0;  af[2 * j + 1] = (_Float16)a1;
      bf[2 * j] = (_Float16)w0;  bf[2 * j + 1] = (_Float16)w1;
    }
    acc = __builtin_amdgcn_wmma_f32_16x16x32_f16(false, af, false, bf,
                                                 (short)0, acc, false, false);
  }

  int nc = nBase + (lane & 15);
  if (nc < Nc) {
    float bv = bias ? bias[nc] : 0.f;
    #pragma unroll
    for (int r = 0; r < 8; ++r) {
      int mr = r + half * 8;
      float v = acc[r] + bv;
      if (add) v += add[mr * Nc + nc];
      if (act == ACT_RELU)      v = fmaxf(v, 0.f);
      else if (act == ACT_SIG)  v = 1.f / (1.f + __expf(-v));
      else if (act == ACT_TANH) v = tanhf(v);
      out[mr * Nc + nc] = v;
    }
  }
}

// --------------------------------------------------------------- max-pool 4x4
// pad 1 with -inf, window 4, stride 4 (VALID on padded)
__global__ void maxpool4_k(const _Float16* __restrict__ in, _Float16* __restrict__ out,
                           int Nb, int C, int H, int W, int Ho, int Wo)
{
  size_t idx = (size_t)blockIdx.x * blockDim.x + threadIdx.x;
  size_t tot = (size_t)Nb * C * Ho * Wo;
  if (idx >= tot) return;
  int wo = idx % Wo; size_t t = idx / Wo;
  int ho = t % Ho;   t /= Ho;
  int c  = t % C;    int n = t / C;
  int h0 = ho * 4 - 1, w0 = wo * 4 - 1;
  float mx = -3.402823e38f;
  for (int dh = 0; dh < 4; ++dh) {
    int h = h0 + dh; if (h < 0 || h >= H) continue;
    for (int dw = 0; dw < 4; ++dw) {
      int w = w0 + dw; if (w < 0 || w >= W) continue;
      float v = (float)in[((size_t)(n * C + c) * H + h) * W + w];
      mx = fmaxf(mx, v);
    }
  }
  out[idx] = (_Float16)mx;
}

// ------------------------------------------------------------------ batchnorm
__global__ __launch_bounds__(256)
void bn_stats_k(const _Float16* __restrict__ x, float* __restrict__ mean,
                float* __restrict__ invstd, int Nb, int C, int HW)
{
  int c = blockIdx.x;
  float s = 0.f, s2 = 0.f;
  int tot = Nb * HW;
  for (int i = threadIdx.x; i < tot; i += 256) {
    int n = i / HW, r = i % HW;
    float v = (float)x[((size_t)(n * C + c)) * HW + r];
    s += v; s2 += v * v;
  }
  __shared__ float sh[256], sh2[256];
  sh[threadIdx.x] = s; sh2[threadIdx.x] = s2;
  __syncthreads();
  for (int st = 128; st > 0; st >>= 1) {
    if (threadIdx.x < st) { sh[threadIdx.x] += sh[threadIdx.x + st];
                            sh2[threadIdx.x] += sh2[threadIdx.x + st]; }
    __syncthreads();
  }
  if (threadIdx.x == 0) {
    float mval = sh[0] / tot;
    float var = sh2[0] / tot - mval * mval;   // biased var, matches jnp.var
    mean[c] = mval;
    invstd[c] = rsqrtf(var + 1e-5f);
  }
}

__global__ void bn_apply_k(_Float16* x, const float* __restrict__ mean,
                           const float* __restrict__ invstd, const float* __restrict__ g,
                           const float* __restrict__ b, int C, int HW, size_t tot)
{
  size_t i = (size_t)blockIdx.x * blockDim.x + threadIdx.x;
  if (i >= tot) return;
  int c = (int)((i / HW) % C);
  float v = (float)x[i];
  x[i] = (_Float16)((v - mean[c]) * invstd[c] * g[c] + b[c]);
}

// ---------------------------------------------------------------- STN sampler
// fI: f16 (B, D, 24, 16); M: f32 (B,6); fsamp: f32 (B, 384, D)  (== Xt layout)
__global__ __launch_bounds__(256)
void stn_k(const _Float16* __restrict__ fI, const float* __restrict__ M,
           float* __restrict__ fsamp, int B_, int D_, int Hh, int Ww)
{
  int b   = blockIdx.x / (Hh * Ww);
  int pos = blockIdx.x % (Hh * Ww);
  int y = pos / Ww, x = pos % Ww;
  const float* Mb = M + b * 6;
  float xs = (2.f * x + 1.f) / Ww - 1.f;
  float ys = (2.f * y + 1.f) / Hh - 1.f;
  float gx = Mb[0] * xs + Mb[1] * ys + Mb[2];
  float gy = Mb[3] * xs + Mb[4] * ys + Mb[5];
  float ix = ((gx + 1.f) * Ww - 1.f) * 0.5f;
  float iy = ((gy + 1.f) * Hh - 1.f) * 0.5f;
  int x0 = (int)floorf(ix), y0 = (int)floorf(iy);
  int x1 = x0 + 1, y1 = y0 + 1;
  float wx1 = ix - x0, wx0 = 1.f - wx1;
  float wy1 = iy - y0, wy0 = 1.f - wy1;
  bool vx0 = (x0 >= 0 && x0 < Ww), vx1 = (x1 >= 0 && x1 < Ww);
  bool vy0 = (y0 >= 0 && y0 < Hh), vy1 = (y1 >= 0 && y1 < Hh);
  int cx0 = min(max(x0, 0), Ww - 1), cx1 = min(max(x1, 0), Ww - 1);
  int cy0 = min(max(y0, 0), Hh - 1), cy1 = min(max(y1, 0), Hh - 1);
  float w00 = wy0 * wx0, w01 = wy0 * wx1, w10 = wy1 * wx0, w11 = wy1 * wx1;

  for (int d = threadIdx.x; d < D_; d += 256) {
    const _Float16* base = fI + ((size_t)(b * D_ + d)) * Hh * Ww;
    float v00 = (vy0 && vx0) ? (float)base[cy0 * Ww + cx0] : 0.f;
    float v01 = (vy0 && vx1) ? (float)base[cy0 * Ww + cx1] : 0.f;
    float v10 = (vy1 && vx0) ? (float)base[cy1 * Ww + cx0] : 0.f;
    float v11 = (vy1 && vx1) ? (float)base[cy1 * Ww + cx1] : 0.f;
    fsamp[((size_t)b * (Hh * Ww) + pos) * D_ + d] =
        v00 * w00 + v01 * w01 + v10 * w10 + v11 * w11;
  }
}

// ----------------------------------------------------------- soft-assign encode
// one wave per (b, n): lane k computes softmax_k(escale[k] * d2[b,n,k])
__global__ __launch_bounds__(32)
void enc_dist_k(const float* __restrict__ Xt, const float* __restrict__ cw,
                const float* __restrict__ escale, float* __restrict__ Aout,
                int Npos, int D_, int K_)
{
  int b = blockIdx.x / Npos, n = blockIdx.x % Npos;
  int k = threadIdx.x;   // K_ == 32 == wave32
  const float* xr = Xt + ((size_t)b * Npos + n) * D_;
  const float* cr = cw + k * D_;
  float dot = 0.f, xn = 0.f, cn = 0.f;
  for (int d = 0; d < D_; ++d) {
    float xv = xr[d], cv = cr[d];
    dot += xv * cv; xn += xv * xv; cn += cv * cv;
  }
  float lg = escale[k] * (xn + cn - 2.f * dot);
  float mx = lg;
  for (int off = 16; off; off >>= 1) mx = fmaxf(mx, __shfl_xor(mx, off, 32));
  float e = __expf(lg - mx);
  float sm = e;
  for (int off = 16; off; off >>= 1) sm += __shfl_xor(sm, off, 32);
  Aout[((size_t)b * Npos + n) * K_ + k] = e / sm;
}

// E[b,k,d] = sum_n A[b,n,k]*Xt[b,n,d] - (sum_n A[b,n,k]) * cw[k,d]
__global__ __launch_bounds__(256)
void enc_E_k(const float* __restrict__ Xt, const float* __restrict__ Aenc,
             const float* __restrict__ cw, float* __restrict__ dst,
             int Npos, int D_, int K_)
{
  int b = blockIdx.x / K_, k = blockIdx.x % K_;
  int d = threadIdx.x;
  if (d >= D_) return;
  float acc = 0.f, asum = 0.f;
  for (int n = 0; n < Npos; ++n) {
    float a = Aenc[((size_t)b * Npos + n) * K_ + k];
    acc  += a * Xt[((size_t)b * Npos + n) * D_ + d];
    asum += a;
  }
  dst[(size_t)b * (K_ * D_) + k * D_ + d] = acc - asum * cw[k * D_ + d];
}

// --------------------------------------------------------------- LSTM update
__global__ void cell_update_k(const float* gi, const float* gg, const float* go,
                              const float* gm, float* h, float* c, int n)
{
  int i = blockIdx.x * blockDim.x + threadIdx.x;
  if (i < n) {
    float cx = gg[i] * c[i] + gi[i] * gm[i];
    c[i] = cx;
    h[i] = go[i] * cx;
  }
}

// ------------------------------------------- final scores: max over T, softmax
__global__ __launch_bounds__(64)
void scores_k(const float* __restrict__ sbuf, float* __restrict__ out,
              int T, int B_, int L)
{
  int b = blockIdx.x;
  int j = threadIdx.x;
  __shared__ float sh[64];
  float v = -3.402823e38f;
  if (j < L)
    for (int t = 0; t < T; ++t)
      v = fmaxf(v, sbuf[((size_t)t * B_ + b) * L + j]);
  sh[j] = (j < L) ? v : -3.402823e38f;
  __syncthreads();
  for (int st = 32; st; st >>= 1) {
    if (j < st) sh[j] = fmaxf(sh[j], sh[j + st]);
    __syncthreads();
  }
  float mx = sh[0];
  __syncthreads();
  float e = (j < L) ? __expf(v - mx) : 0.f;
  sh[j] = e;
  __syncthreads();
  for (int st = 32; st; st >>= 1) {
    if (j < st) sh[j] += sh[j + st];
    __syncthreads();
  }
  float sum = sh[0];
  if (j < L) out[b * L + j] = e / sum;
}

// scale_c / pos_c from final M, anch_c summed over M_list (= Marr slots 2..10)
__global__ void scalars_k(const float* __restrict__ Marr, float* out_scale,
                          float* out_pos, float* out_anch, int B_)
{
  int b = threadIdx.x;
  if (b >= B_) return;
  const float* Mf = Marr + (size_t)10 * B_ * 6 + b * 6;
  float a = fmaxf(Mf[0] - 0.5f, 0.f), d = fmaxf(Mf[4] - 0.5f, 0.f);
  out_scale[b] = a * a + d * d;
  out_pos[b] = fmaxf(0.1f - Mf[0], 0.f) + fmaxf(0.1f - Mf[4], 0.f);
  float anch = 0.f;
  for (int t = 2; t <= 10; ++t) {
    const float* Mt = Marr + (size_t)t * B_ * 6 + b * 6;
    float tx = Mt[2] - 0.5f, ty = Mt[5];
    anch += 0.5f * (tx * tx + ty * ty);
  }
  out_anch[b] = anch;
}

// ============================================================================
extern "C" void kernel_launch(void* const* d_in, const int* in_sizes, int n_in,
                              void* d_out, int out_size, void* d_ws, size_t ws_size,
                              hipStream_t stream)
{
  (void)in_sizes; (void)n_in; (void)out_size; (void)ws_size;
  const int B_ = 16, D_ = 256, Kc = 32, HF = 24, WF = 16, L = 59;

  const float* x     = (const float*)d_in[0];
  const float* w11   = (const float*)d_in[1];  const float* b11 = (const float*)d_in[2];
  const float* w12   = (const float*)d_in[3];  const float* b12 = (const float*)d_in[4];
  const float* w21   = (const float*)d_in[5];  const float* b21 = (const float*)d_in[6];
  const float* w22   = (const float*)d_in[7];  const float* b22 = (const float*)d_in[8];
  const float* w31   = (const float*)d_in[9];  const float* b31 = (const float*)d_in[10];
  const float* w32   = (const float*)d_in[11]; const float* b32 = (const float*)d_in[12];
  const float* bn1g  = (const float*)d_in[13]; const float* bn1b = (const float*)d_in[14];
  const float* bn2g  = (const float*)d_in[15]; const float* bn2b = (const float*)d_in[16];
  const float* cw    = (const float*)d_in[17]; const float* esc  = (const float*)d_in[18];
  const float* Wfx   = (const float*)d_in[19]; const float* bfx  = (const float*)d_in[20];
  const float* Wxi   = (const float*)d_in[21]; const float* bxi  = (const float*)d_in[22];
  const float* Whi   = (const float*)d_in[23]; const float* bhi  = (const float*)d_in[24];
  const float* Wxg   = (const float*)d_in[25]; const float* bxg  = (const float*)d_in[26];
  const float* Whg   = (const float*)d_in[27]; const float* bhg  = (const float*)d_in[28];
  const float* Wxo   = (const float*)d_in[29]; const float* bxo  = (const float*)d_in[30];
  const float* Who   = (const float*)d_in[31]; const float* bho  = (const float*)d_in[32];
  const float* Wxm   = (const float*)d_in[33]; const float* bxm  = (const float*)d_in[34];
  const float* Whm   = (const float*)d_in[35]; const float* bhm  = (const float*)d_in[36];
  const float* Whz   = (const float*)d_in[37]; const float* bhz  = (const float*)d_in[38];
  const float* Wzs   = (const float*)d_in[39]; const float* bzs  = (const float*)d_in[40];
  const float* Wzm   = (const float*)d_in[41]; const float* bzm  = (const float*)d_in[42];

  float* out = (float*)d_out;   // [scores 944 | scale 16 | pos 16 | anch 16 | feats 9*16*8192]

  // ---------------- workspace layout ----------------
  char* wbase = (char*)d_ws;
  size_t off = 0;
  auto alloc = [&](size_t bytes) -> char* {
    char* p = wbase + off;
    off += (bytes + 255) & ~(size_t)255;
    return p;
  };
  _Float16* xh    = (_Float16*)alloc((size_t)B_ * 3 * 384 * 256 * 2);
  _Float16* buf0  = (_Float16*)alloc((size_t)B_ * 64 * 384 * 256 * 2);
  _Float16* buf1  = (_Float16*)alloc((size_t)B_ * 64 * 384 * 256 * 2);
  _Float16* fI    = (_Float16*)alloc((size_t)B_ * D_ * HF * WF * 2);
  float* fsamp    = (float*)alloc((size_t)B_ * HF * WF * D_ * 4);
  float* Aenc     = (float*)alloc((size_t)B_ * HF * WF * Kc * 4);
  float* fc0      = (float*)alloc((size_t)B_ * Kc * D_ * 4);
  float* hbuf     = (float*)alloc((size_t)B_ * D_ * 4);
  float* cbuf     = (float*)alloc((size_t)B_ * D_ * 4);
  float* xk       = (float*)alloc((size_t)B_ * D_ * 4);
  float* tmp      = (float*)alloc((size_t)B_ * D_ * 4);
  float* gi       = (float*)alloc((size_t)B_ * D_ * 4);
  float* gg       = (float*)alloc((size_t)B_ * D_ * 4);
  float* go       = (float*)alloc((size_t)B_ * D_ * 4);
  float* gm       = (float*)alloc((size_t)B_ * D_ * 4);
  float* zb       = (float*)alloc((size_t)B_ * D_ * 4);
  float* sscratch = (float*)alloc((size_t)B_ * L * 4);
  float* sbuf     = (float*)alloc((size_t)9 * B_ * L * 4);
  float* Marr     = (float*)alloc((size_t)11 * B_ * 6 * 4);
  float* bnMean   = (float*)alloc(256 * 4);
  float* bnIstd   = (float*)alloc(256 * 4);

  // ---------------- launch helpers ----------------
  auto conv = [&](const _Float16* in, const float* w, const float* bs, _Float16* o,
                  int Cin, int Cout, int H, int W, int relu) {
    int Mtot = B_ * H * W;
    dim3 g((Mtot + 63) / 64, (Cout + 63) / 64);
    conv3x3_wmma<<<g, dim3(256), 0, stream>>>(in, w, bs, o, B_, Cin, Cout, H, W, relu);
  };
  auto g16 = [&](const float* A, const float* W, const float* bs, const float* add,
                 float* o, int K, int N, int act) {
    gemm16_wmma<<<dim3((N + 15) / 16), dim3(32), 0, stream>>>(A, W, bs, add, o, K, N, act);
  };
  auto pool = [&](const _Float16* in, _Float16* o, int C, int H, int W, int Ho, int Wo) {
    size_t tot = (size_t)B_ * C * Ho * Wo;
    maxpool4_k<<<dim3((unsigned)((tot + 255) / 256)), dim3(256), 0, stream>>>(in, o, B_, C, H, W, Ho, Wo);
  };
  auto bn = [&](_Float16* buf, const float* g, const float* bb, int C, int H, int W) {
    bn_stats_k<<<dim3(C), dim3(256), 0, stream>>>(buf, bnMean, bnIstd, B_, C, H * W);
    size_t tot = (size_t)B_ * C * H * W;
    bn_apply_k<<<dim3((unsigned)((tot + 255) / 256)), dim3(256), 0, stream>>>(
        buf, bnMean, bnIstd, g, bb, C, H * W, tot);
  };

  // ---------------- CNN backbone ----------------
  {
    size_t nx = (size_t)B_ * 3 * 384 * 256;
    cvt_f16_k<<<dim3((unsigned)((nx + 255) / 256)), dim3(256), 0, stream>>>(x, xh, nx);
  }
  conv(xh,   w11, b11, buf0, 3,   64,  384, 256, 1);   // conv11 + relu
  conv(buf0, w12, b12, buf1, 64,  64,  384, 256, 1);   // conv12 + relu
  pool(buf1, buf0, 64, 384, 256, 96, 64);              // maxpool4 -> 96x64
  bn(buf0, bn1g, bn1b, 64, 96, 64);                    // bn1 in-place
  conv(buf0, w21, b21, buf1, 64,  128, 96, 64, 1);     // conv21 + relu
  conv(buf1, w22, b22, buf0, 128, 128, 96, 64, 1);     // conv22 + relu
  pool(buf0, buf1, 128, 96, 64, 24, 16);               // maxpool4 -> 24x16
  bn(buf1, bn2g, bn2b, 128, 24, 16);                   // bn2 in-place
  conv(buf1, w31, b31, buf0, 128, 256, 24, 16, 1);     // conv31 + relu
  conv(buf0, w32, b32, fI,   256, 256, 24, 16, 0);     // conv32 (no relu) -> f_I

  // ---------------- recurrent glimpse loop ----------------
  init_M_k<<<dim3(1), dim3(128), 0, stream>>>(Marr, B_);        // Marr[0] = identity
  zero_f32_k<<<dim3(16), dim3(256), 0, stream>>>(hbuf, B_ * D_);
  zero_f32_k<<<dim3(16), dim3(256), 0, stream>>>(cbuf, B_ * D_);

  const int Npos = HF * WF;  // 384
  auto glimpse = [&](const float* Mptr, float* fcdst) {
    stn_k<<<dim3(B_ * Npos), dim3(256), 0, stream>>>(fI, Mptr, fsamp, B_, D_, HF, WF);
    enc_dist_k<<<dim3(B_ * Npos), dim3(32), 0, stream>>>(fsamp, cw, esc, Aenc, Npos, D_, Kc);
    enc_E_k<<<dim3(B_ * Kc), dim3(256), 0, stream>>>(fsamp, Aenc, cw, fcdst, Npos, D_, Kc);
  };
  auto cell = [&](const float* fc, float* sdst, float* Mdst) {
    g16(fc,   Wfx, bfx, nullptr, xk,  Kc * D_, D_, ACT_RELU);   // xk
    g16(hbuf, Whi, bhi, nullptr, tmp, D_, D_, ACT_NONE);
    g16(xk,   Wxi, bxi, tmp,     gi,  D_, D_, ACT_SIG);         // i
    g16(hbuf, Whg, bhg, nullptr, tmp, D_, D_, ACT_NONE);
    g16(xk,   Wxg, bxg, tmp,     gg,  D_, D_, ACT_SIG);         // g
    g16(hbuf, Who, bho, nullptr, tmp, D_, D_, ACT_NONE);
    g16(xk,   Wxo, bxo, tmp,     go,  D_, D_, ACT_SIG);         // o
    g16(hbuf, Whm, bhm, nullptr, tmp, D_, D_, ACT_NONE);
    g16(xk,   Wxm, bxm, tmp,     gm,  D_, D_, ACT_TANH);        // m
    cell_update_k<<<dim3(16), dim3(256), 0, stream>>>(gi, gg, go, gm, hbuf, cbuf, B_ * D_);
    g16(hbuf, Whz, bhz, nullptr, zb, D_, D_, ACT_RELU);         // z
    g16(zb,   Wzs, bzs, nullptr, sdst, D_, L, ACT_NONE);        // s
    g16(zb,   Wzm, bzm, nullptr, Mdst, D_, 6, ACT_NONE);        // M
  };

  // warm-up step (s discarded; M -> Marr[1])
  glimpse(Marr, fc0);
  cell(fc0, sscratch, Marr + (size_t)1 * B_ * 6);

  // 9 recorded steps: features -> d_out, s -> sbuf[t], M -> Marr[t+2]
  for (int t = 0; t < 9; ++t) {
    float* fcdst = out + 992 + (size_t)t * B_ * Kc * D_;
    glimpse(Marr + (size_t)(t + 1) * B_ * 6, fcdst);
    cell(fcdst, sbuf + (size_t)t * B_ * L, Marr + (size_t)(t + 2) * B_ * 6);
  }

  // ---------------- outputs ----------------
  scores_k<<<dim3(B_), dim3(64), 0, stream>>>(sbuf, out, 9, B_, L);
  scalars_k<<<dim3(1), dim3(32), 0, stream>>>(Marr, out + 944, out + 960, out + 976, B_);
}
